// GCNEncoder_24893630448153
// MI455X (gfx1250) — matrix operand order repaired and checked by reading, not compiled, not run on hardware
//
#include <hip/hip_runtime.h>

// Problem constants (match reference)
#define Uh   256      // hidden size
#define Tn   128      // sequence length
#define Bn   128      // batch
#define BT   (Bn*Tn)  // 16384 nodes
#define DEGk 5
#define NLk  64
#define Ek   (BT*DEGk) // 81920 edges

typedef __attribute__((ext_vector_type(16))) __bf16 v16bf;
typedef __attribute__((ext_vector_type(8)))  __bf16 v8bf;
typedef __attribute__((ext_vector_type(8)))  float  v8f;

static __device__ __forceinline__ __bf16 tobf(float f) { return (__bf16)f; }
static __device__ __forceinline__ float sigmoid_f(float x) { return 1.0f / (1.0f + __expf(-x)); }

// ---------------------------------------------------------------------------
// Embedding gather: x[n=b*T+t, u] = emb[src[t,b], u]
// ---------------------------------------------------------------------------
__global__ void embed_kernel(const int* __restrict__ src, const float* __restrict__ emb,
                             float* __restrict__ x) {
    int n = blockIdx.x, u = threadIdx.x;
    int b = n / Tn, t = n % Tn;
    int tok = src[t * Bn + b];
    x[n * Uh + u] = emb[tok * Uh + u];
}

// ---------------------------------------------------------------------------
// f32 -> bf16 conversion (activations / enc), one element per thread
// ---------------------------------------------------------------------------
__global__ void cvt_bf16_kernel(const float* __restrict__ x, __bf16* __restrict__ xb) {
    size_t i = (size_t)blockIdx.x * blockDim.x + threadIdx.x;
    xb[i] = tobf(x[i]);
}

// ---------------------------------------------------------------------------
// Pack one 256x256 f32 weight matrix into WMMA-B fragment order (bf16).
// Fragment (nt, kb) for lane l: b[2v+p] = Bw[(kb*32 + hi*16 + 2v + p)*256 + nt*16+lo]
// Stored contiguously: Bp[((nt*8 + kb)*32 + lane)*16 + e]  (32 bytes/lane-fragment)
// ---------------------------------------------------------------------------
__global__ void pack_b_kernel(const float* __restrict__ Bw, __bf16* __restrict__ Bp) {
    int tid  = blockIdx.x * blockDim.x + threadIdx.x;   // = nt*256 + kb*32 + lane
    int lane = tid & 31;
    int kb   = (tid >> 5) & 7;
    int hi   = lane >> 4, lo = lane & 15;
    int nt   = tid >> 8;
    const float* bcol = Bw + (kb * 32 + hi * 16) * Uh + nt * 16 + lo;
    __bf16* dst = Bp + (size_t)tid * 16;
    #pragma unroll
    for (int v = 0; v < 8; ++v) {
        dst[2 * v]     = tobf(bcol[(2 * v) * Uh]);
        dst[2 * v + 1] = tobf(bcol[(2 * v + 1) * Uh]);
    }
}

// ---------------------------------------------------------------------------
// GEMM C[M,256] = A[M,256] * B[256,256] with v_wmma_f32_16x16x32_bf16.
// A: bf16 row-major. Bp: fragment-packed bf16 (see pack_b_kernel).
// Block = 256 threads = 8 waves. The block's 64-column B strip (32 KB,
// contiguous in Bp) is staged global->LDS once with
// GLOBAL_LOAD_ASYNC_TO_LDS_B128 (ASYNCcnt), then all 8 waves read their
// fragments from LDS with immediate-offset ds_load_b128.
// Each wave computes a 16x64 C strip: 4 N-tiles x K fully unrolled = 32 WMMA.
// grid.x = N/64 = 4, grid.y = M/128.
// ---------------------------------------------------------------------------
__global__ void gemm_wmma_bf16(const __bf16* __restrict__ A, const __bf16* __restrict__ Bp,
                               float* __restrict__ C) {
    __shared__ __bf16 sB[4 * 8 * 32 * 16];   // 16384 bf16 = 32 KB

    const int tid  = threadIdx.x;
    const int wave = tid >> 5;
    const int lane = tid & 31;
    const int hi   = lane >> 4;
    const int lo   = lane & 15;
    const int mt   = blockIdx.y * 8 + wave;   // 16-row tile index
    const int mrow = mt * 16 + lo;

    // ---- Async stage of this block's packed-B strip (32 KB) into LDS ----
    {
        const char* bsrc = (const char*)(Bp + (size_t)blockIdx.x * (4 * 8 * 32 * 16));
        unsigned ldsBase = (unsigned)(size_t)(&sB[0]);
        #pragma unroll
        for (int it = 0; it < 8; ++it) {
            unsigned off = (unsigned)(it * 256 + tid) * 16u;   // 16B chunk per thread
            unsigned ldsAddr = ldsBase + off;
            const char* g = bsrc + off;
            asm volatile("global_load_async_to_lds_b128 %0, %1, off"
                         :: "v"(ldsAddr), "v"(g)
                         : "memory");
        }
        asm volatile("s_wait_asynccnt 0x0" ::: "memory");
    }
    __syncthreads();

    const __bf16* arow  = A + (size_t)mrow * Uh + hi * 16;  // per-lane A base
    const __bf16* sbase = sB + lane * 16;                    // per-lane LDS fragment base

    v8f c0 = {}, c1 = {}, c2 = {}, c3 = {};

    #pragma unroll
    for (int kb = 0; kb < 8; ++kb) {
        // A fragment: two contiguous 16B chunks per lane (immediate offsets)
        v8bf a_lo = *(const v8bf*)(arow + kb * 32);
        v8bf a_hi = *(const v8bf*)(arow + kb * 32 + 8);
        v16bf a = __builtin_shufflevector(a_lo, a_hi,
                                          0, 1, 2, 3, 4, 5, 6, 7,
                                          8, 9, 10, 11, 12, 13, 14, 15);
        // 4 B fragments from LDS (32B each, immediate ds offsets)
        const v16bf b0 = *(const v16bf*)(sbase + (0 * 8 + kb) * 512);
        const v16bf b1 = *(const v16bf*)(sbase + (1 * 8 + kb) * 512);
        const v16bf b2 = *(const v16bf*)(sbase + (2 * 8 + kb) * 512);
        const v16bf b3 = *(const v16bf*)(sbase + (3 * 8 + kb) * 512);
        c0 = __builtin_amdgcn_wmma_f32_16x16x32_bf16(false, a, false, b0, (short)0, c0, false, false);
        c1 = __builtin_amdgcn_wmma_f32_16x16x32_bf16(false, a, false, b1, (short)0, c1, false, false);
        c2 = __builtin_amdgcn_wmma_f32_16x16x32_bf16(false, a, false, b2, (short)0, c2, false, false);
        c3 = __builtin_amdgcn_wmma_f32_16x16x32_bf16(false, a, false, b3, (short)0, c3, false, false);
    }

    // C/D layout: VGPR r -> M = r + 8*hi, N = lane&15 (immediate store offsets)
    float* cbase = C + (size_t)(mt * 16 + hi * 8) * Uh + blockIdx.x * 64 + lo;
    #pragma unroll
    for (int r = 0; r < 8; ++r) {
        cbase[r * Uh +  0] = c0[r];
        cbase[r * Uh + 16] = c1[r];
        cbase[r * Uh + 32] = c2[r];
        cbase[r * Uh + 48] = c3[r];
    }
}

// ---------------------------------------------------------------------------
// Gating GEMVs: g*(n) = sum_u x[n,u] * vg*(u)  (three at once, LDS reduce)
// ---------------------------------------------------------------------------
__global__ void gates_kernel(const float* __restrict__ x,
                             const float* __restrict__ vg_in, const float* __restrict__ vg_out,
                             const float* __restrict__ wg_loop,
                             float* __restrict__ gin, float* __restrict__ gout,
                             float* __restrict__ gloop) {
    __shared__ float s0[256], s1[256], s2[256];
    int n = blockIdx.x, u = threadIdx.x;
    float xv = x[n * Uh + u];
    s0[u] = xv * vg_in[u];
    s1[u] = xv * vg_out[u];
    s2[u] = xv * wg_loop[u];
    __syncthreads();
    for (int s = 128; s > 0; s >>= 1) {
        if (u < s) { s0[u] += s0[u + s]; s1[u] += s1[u + s]; s2[u] += s2[u + s]; }
        __syncthreads();
    }
    if (u == 0) { gin[n] = s0[0]; gout[n] = s1[0]; gloop[n] = s2[0]; }
}

// ---------------------------------------------------------------------------
// Edge gather + gated sum + ReLU + sentence mask.
// ---------------------------------------------------------------------------
__global__ void combine_kernel(const float* __restrict__ Yin, const float* __restrict__ Yout,
                               const float* __restrict__ Yloop,
                               const float* __restrict__ gin, const float* __restrict__ gout,
                               const float* __restrict__ gloop,
                               const float* __restrict__ b_in, const float* __restrict__ bg_in,
                               const float* __restrict__ b_out, const float* __restrict__ bg_out,
                               const int* __restrict__ arc_in, const int* __restrict__ lab_in,
                               const int* __restrict__ arc_out, const int* __restrict__ lab_out,
                               const float* __restrict__ m_in, const float* __restrict__ m_out,
                               const float* __restrict__ m_loop, const float* __restrict__ sent,
                               float* __restrict__ xo) {
    int n = blockIdx.x, u = threadIdx.x;
    float acc = 0.0f;
    #pragma unroll
    for (int k = 0; k < DEGk; ++k) {
        int e  = n * DEGk + k;
        int s  = arc_in[e] * Tn + arc_in[Ek + e];   // idx_in = arc[0]*T + arc[1]
        int li = lab_in[e];
        float p = sigmoid_f(gin[s] + bg_in[li]) * m_in[e];
        acc += p * (Yin[s * Uh + u] + b_in[li * Uh + u]);
    }
    #pragma unroll
    for (int k = 0; k < DEGk; ++k) {
        int e  = n * DEGk + k;
        int s  = arc_out[e] * Tn + arc_out[Ek + e];
        int li = lab_out[e];
        float p = sigmoid_f(gout[s] + bg_out[li]) * m_out[e];
        acc += p * (Yout[s * Uh + u] + b_out[li * Uh + u]);
    }
    float pl = sigmoid_f(gloop[n]) * m_loop[n];
    acc += pl * Yloop[n * Uh + u];
    int b = n / Tn, t = n % Tn;
    xo[n * Uh + u] = fmaxf(acc, 0.0f) * sent[t * Bn + b];
}

// ---------------------------------------------------------------------------
// enc[b,u] = (sum_t x[b*T+t, u]) / (sum_t sent[t,b])
// ---------------------------------------------------------------------------
__global__ void enc_kernel(const float* __restrict__ x, const float* __restrict__ sent,
                           float* __restrict__ enc) {
    __shared__ float msum;
    int b = blockIdx.x, u = threadIdx.x;
    if (u == 0) {
        float s = 0.0f;
        for (int t = 0; t < Tn; ++t) s += sent[t * Bn + b];
        msum = s;
    }
    __syncthreads();
    float acc = 0.0f;
    for (int t = 0; t < Tn; ++t) acc += x[(b * Tn + t) * Uh + u];
    enc[b * Uh + u] = acc / msum;
}

// ---------------------------------------------------------------------------
// memory_bank: [BT,U] node-major (n=b*T+t) -> [T,B,U]
// ---------------------------------------------------------------------------
__global__ void bank_kernel(const float* __restrict__ x, float* __restrict__ out) {
    int n = blockIdx.x, u = threadIdx.x;
    int b = n / Tn, t = n % Tn;
    out[(t * Bn + b) * Uh + u] = x[n * Uh + u];
}

// ---------------------------------------------------------------------------
extern "C" void kernel_launch(void* const* d_in, const int* in_sizes, int n_in,
                              void* d_out, int out_size, void* d_ws, size_t ws_size,
                              hipStream_t stream) {
    const int*   src     = (const int*)  d_in[0];
    const int*   arc_in  = (const int*)  d_in[1];
    const int*   arc_out = (const int*)  d_in[2];
    const int*   lab_in  = (const int*)  d_in[3];
    const int*   lab_out = (const int*)  d_in[4];
    const float* m_in    = (const float*)d_in[5];
    const float* m_out   = (const float*)d_in[6];
    const float* m_loop  = (const float*)d_in[7];
    const float* sent    = (const float*)d_in[8];
    const float* emb     = (const float*)d_in[9];
    const float* V_in    = (const float*)d_in[10];
    const float* b_in    = (const float*)d_in[11];
    const float* Vg_in   = (const float*)d_in[12];
    const float* bg_in   = (const float*)d_in[13];
    const float* V_out   = (const float*)d_in[14];
    const float* b_out   = (const float*)d_in[15];
    const float* Vg_out  = (const float*)d_in[16];
    const float* bg_out  = (const float*)d_in[17];
    const float* W_loop  = (const float*)d_in[18];
    const float* Wg_loop = (const float*)d_in[19];
    const float* H       = (const float*)d_in[20];
    float* out = (float*)d_out;

    // ---- Workspace layout ----
    float* w     = (float*)d_ws;
    float* xA    = w;                            // [BT,U] f32
    float* xB    = xA    + (size_t)BT * Uh;
    float* Yin   = xB    + (size_t)BT * Uh;
    float* Yout  = Yin   + (size_t)BT * Uh;
    float* Yloop = Yout  + (size_t)BT * Uh;
    float* gin   = Yloop + (size_t)BT * Uh;
    float* gout  = gin   + BT;
    float* gloop = gout  + BT;
    float* encb  = gloop + BT;                   // [B,U] f32
    __bf16* xbf   = (__bf16*)(encb + (size_t)Bn * Uh);   // [BT,U] bf16
    __bf16* encbf = xbf + (size_t)BT * Uh;               // [B,U]  bf16
    __bf16* Bp    = encbf + (size_t)Bn * Uh;             // 10 packed 256x256 matrices
    const size_t WSLOT = (size_t)Uh * Uh;                // 65536 bf16 per matrix

    // ---- Pack all weight matrices into WMMA-B fragment order (bf16) ----
    // slots: 0..2 layer0 {V_in, V_out, W_loop}; 3..5 layer1; 6..9 H[0..3]
    for (int l = 0; l < 2; ++l) {
        pack_b_kernel<<<16, 256, 0, stream>>>(V_in  + (size_t)l * Uh * Uh, Bp + (size_t)(l * 3 + 0) * WSLOT);
        pack_b_kernel<<<16, 256, 0, stream>>>(V_out + (size_t)l * Uh * Uh, Bp + (size_t)(l * 3 + 1) * WSLOT);
        pack_b_kernel<<<16, 256, 0, stream>>>(W_loop + (size_t)l * Uh * Uh, Bp + (size_t)(l * 3 + 2) * WSLOT);
    }
    for (int k = 0; k < 4; ++k)
        pack_b_kernel<<<16, 256, 0, stream>>>(H + (size_t)k * Uh * Uh, Bp + (size_t)(6 + k) * WSLOT);

    dim3 gBig(Uh / 64, BT / 128);   // (4, 128) for M=16384 GEMMs, 256-thread blocks
    dim3 gEnc(Uh / 64, Bn / 128);   // (4, 1)   for M=128 GEMMs

    embed_kernel<<<BT, 256, 0, stream>>>(src, emb, xA);

    for (int l = 0; l < 2; ++l) {
        const float* xin = (l == 0) ? xA : xB;
        float*       xo  = (l == 0) ? xB : xA;
        cvt_bf16_kernel<<<(BT * Uh) / 256, 256, 0, stream>>>(xin, xbf);
        gemm_wmma_bf16<<<gBig, 256, 0, stream>>>(xbf, Bp + (size_t)(l * 3 + 0) * WSLOT, Yin);
        gemm_wmma_bf16<<<gBig, 256, 0, stream>>>(xbf, Bp + (size_t)(l * 3 + 1) * WSLOT, Yout);
        gemm_wmma_bf16<<<gBig, 256, 0, stream>>>(xbf, Bp + (size_t)(l * 3 + 2) * WSLOT, Yloop);
        gates_kernel<<<BT, 256, 0, stream>>>(xin, Vg_in + l * Uh, Vg_out + l * Uh,
                                             Wg_loop + l * Uh, gin, gout, gloop);
        combine_kernel<<<BT, 256, 0, stream>>>(Yin, Yout, Yloop, gin, gout, gloop,
                                               b_in  + l * NLk * Uh, bg_in  + l * NLk,
                                               b_out + l * NLk * Uh, bg_out + l * NLk,
                                               arc_in, lab_in, arc_out, lab_out,
                                               m_in, m_out, m_loop, sent, xo);
    }
    // final x lives in xA after layer 1
    enc_kernel<<<Bn, 256, 0, stream>>>(xA, sent, encb);
    cvt_bf16_kernel<<<(Bn * Uh) / 256, 256, 0, stream>>>(encb, encbf);
    for (int k = 0; k < 4; ++k) {
        gemm_wmma_bf16<<<gEnc, 256, 0, stream>>>(encbf, Bp + (size_t)(6 + k) * WSLOT,
                                                 out + (size_t)k * Bn * Uh);
    }
    bank_kernel<<<BT, 256, 0, stream>>>(xA, out + (size_t)4 * Bn * Uh);
}
